// RandAugment_18184891531907
// MI455X (gfx1250) — compile-verified, matrix operand order checked by reading, not run on hardware
//
#include <hip/hip_runtime.h>
#include <math.h>

// RandAugment (B=64, C=3, H=W=224, DEPTH=2) for gfx1250 (MI455X).
//
// Roofline: x + tmp + out ~= 115MB which FITS in the 192MB global L2, so HBM
// traffic is ~one cold read of x + writeback of out (~77MB -> ~3.3us at
// 23.3TB/s); the rest is L2-resident. That makes wide b128 vector accesses
// (not NT hints) the right data-movement choice -> all streaming ops use
// float4. One block per image fuses both depth passes (x -> ws -> out) with a
// block barrier; whole-image stats are block-local (wave32 shuffles + LDS).
// Luminance dot (color/contrast) runs on v_wmma_f32_16x16x32_f16 with an
// error-compensated f16 split (f32-grade accuracy).

#define C_ 3
#define H_ 224
#define W_ 224
#define NT_MAIN 512

constexpr int kHW  = H_ * W_;   // 50176 (divisible by 16)
constexpr int kCHW = C_ * kHW;  // 150528 (divisible by 4)

typedef __attribute__((ext_vector_type(16))) _Float16 v16h;
typedef __attribute__((ext_vector_type(8)))  float    v8f;

__device__ __forceinline__ void assume_global(const void* p) {
#if __has_builtin(__builtin_amdgcn_is_shared) && __has_builtin(__builtin_amdgcn_is_private)
  __builtin_assume(!__builtin_amdgcn_is_shared((void*)p));
  __builtin_assume(!__builtin_amdgcn_is_private((void*)p));
#endif
}

__device__ __forceinline__ float clip255(float v) { return fminf(fmaxf(v, 0.0f), 255.0f); }
__device__ __forceinline__ float linf(float m, float lo, float hi) { return lo + m * (hi - lo); }

struct Src { const float* p; float sc; };
__device__ __forceinline__ float ldi(const Src& s, int i) { return s.p[i] * s.sc; }
__device__ __forceinline__ float ldpx(const Src& s, int c, int y, int x) {
  return s.p[c * kHW + y * W_ + x] * s.sc;
}
__device__ __forceinline__ void stv(float* d, int i, float v, bool fin) {
  if (fin) v = fminf(fmaxf(v / 255.0f, 0.0f), 1.0f);  // final clip(x/255, 0, 1)
  d[i] = v;
}
__device__ __forceinline__ float4 fin4(float4 v, bool fin) {
  if (fin) {
    v.x = fminf(fmaxf(v.x / 255.0f, 0.0f), 1.0f);
    v.y = fminf(fmaxf(v.y / 255.0f, 0.0f), 1.0f);
    v.z = fminf(fmaxf(v.z / 255.0f, 0.0f), 1.0f);
    v.w = fminf(fmaxf(v.w / 255.0f, 0.0f), 1.0f);
  }
  return v;
}

// Vectorized elementwise map over a contiguous range of n4 float4's.
template<typename F>
__device__ __forceinline__ void stream_range(const float* sp_, float sc, float* dp_,
                                             int n4, bool fin, int tid, int nt, F fn) {
  const float4* sp = (const float4*)sp_;
  float4* dp = (float4*)dp_;
  for (int i = tid; i < n4; i += nt) {
    float4 v = sp[i];
    float4 o;
    o.x = fn(v.x * sc); o.y = fn(v.y * sc); o.z = fn(v.z * sc); o.w = fn(v.w * sc);
    dp[i] = fin4(o, fin);
  }
}

// ---------------- wave32 / block reductions ----------------
template<int OP>  // 0=sum 1=min 2=max
__device__ __forceinline__ float warp_red(float v) {
#pragma unroll
  for (int o = 16; o > 0; o >>= 1) {
    float t = __shfl_down(v, o, 32);
    if (OP == 0) v = v + t;
    else if (OP == 1) v = fminf(v, t);
    else v = fmaxf(v, t);
  }
  return v;
}

template<int OP>
__device__ float block_red(float v, float* s, int tid) {
  const int lane = tid & 31, wv = tid >> 5, nw = (int)(blockDim.x >> 5);
  v = warp_red<OP>(v);
  if (lane == 0) s[wv] = v;
  __syncthreads();
  if (wv == 0) {
    float t = (lane < nw) ? s[lane] : (OP == 0 ? 0.0f : s[0]);
    t = warp_red<OP>(t);
    if (lane == 0) s[0] = t;
  }
  __syncthreads();
  float r = s[0];
  __syncthreads();
  return r;
}

// ---------------- WMMA luminance: 16 pixels/wave per issue ----------------
// gray = 0.299R + 0.587G + 0.114B as D = A(16x32 f16) x B(32x16 f16):
// K=0..2: wh_c * xh_c ; K=3..5: wh_c * (xl_c + (wl_c/wh_c)*xh_c) -- the f16
// weight residual is folded into the pixel low part, so the result matches
// the f32 dot to ~2e-5 abs on a 0..255 scale. All rows of A are identical,
// so every D row holds gray(pixel N); lane L reads d[0] = gray(pixel L%16).
__device__ __forceinline__ float wave_gray16(float R, float G, float Bv, int lane) {
  const float wRf = 0.299f, wGf = 0.587f, wBf = 0.114f;
  const _Float16 whR = (_Float16)wRf, whG = (_Float16)wGf, whB = (_Float16)wBf;
  const float rR = (wRf - (float)whR) / (float)whR;
  const float rG = (wGf - (float)whG) / (float)whG;
  const float rB = (wBf - (float)whB) / (float)whB;
  v16h a = {};
  v16h b = {};
  v8f  c = {};
  if (lane < 16) {
    a[0] = whR; a[1] = whG; a[2] = whB;     // K=0..2 (lanes 0-15 hold K<8)
    a[3] = whR; a[4] = whG; a[5] = whB;     // K=3..5
    _Float16 xhR = (_Float16)R, xhG = (_Float16)G, xhB = (_Float16)Bv;
    b[0] = xhR; b[1] = xhG; b[2] = xhB;
    b[3] = (_Float16)((R  - (float)xhR) + rR * (float)xhR);
    b[4] = (_Float16)((G  - (float)xhG) + rG * (float)xhG);
    b[5] = (_Float16)((Bv - (float)xhB) + rB * (float)xhB);
  }
  c = __builtin_amdgcn_wmma_f32_16x16x32_f16(false, a, false, b, (short)0, c, false, false);
  return c[0];
}

// ---------------- ops ----------------
__device__ __forceinline__ float tap(const Src& s, int c, float yf, float xf) {
  bool valid = (xf >= 0.0f) && (xf <= (float)(W_ - 1)) &&
               (yf >= 0.0f) && (yf <= (float)(H_ - 1));
  int xi = (int)fminf(fmaxf(xf, 0.0f), (float)(W_ - 1));
  int yi = (int)fminf(fmaxf(yf, 0.0f), (float)(H_ - 1));
  float v = ldpx(s, c, yi, xi);
  return valid ? v : 0.0f;
}

__device__ void op_affine(const Src& s, float* dst, bool fin,
                          float a, float b, float c, float d, float e, float f,
                          int tid, int nt) {
  for (int i = tid; i < kHW; i += nt) {
    int y = i / W_, x = i - y * W_;
    float xs = (float)x, ys = (float)y;
    float sx = a * xs + b * ys + c;
    float sy = d * xs + e * ys + f;
    float x0 = floorf(sx), y0 = floorf(sy);
    float wx = sx - x0, wy = sy - y0;
#pragma unroll
    for (int ch = 0; ch < C_; ++ch) {
      float v00 = tap(s, ch, y0, x0);
      float v01 = tap(s, ch, y0, x0 + 1.0f);
      float v10 = tap(s, ch, y0 + 1.0f, x0);
      float v11 = tap(s, ch, y0 + 1.0f, x0 + 1.0f);
      float top = v00 * (1.0f - wx) + v01 * wx;
      float bot = v10 * (1.0f - wx) + v11 * wx;
      stv(dst, ch * kHW + i, top * (1.0f - wy) + bot * wy, fin);
    }
  }
}

__device__ void op_color(const Src& s, float* dst, bool fin, float f, int tid) {
  const int lane = tid & 31, wv = tid >> 5, nw = (int)(blockDim.x >> 5);
  for (int c0 = wv; c0 < kHW / 16; c0 += nw) {
    int p = c0 * 16 + (lane & 15);
    float R = 0.0f, G = 0.0f, Bv = 0.0f;
    if (lane < 16) { R = ldi(s, p); G = ldi(s, kHW + p); Bv = ldi(s, 2 * kHW + p); }
    float g = wave_gray16(R, G, Bv, lane);
    if (lane < 16) {
      stv(dst, p,            clip255(g + f * (R  - g)), fin);
      stv(dst, kHW + p,      clip255(g + f * (G  - g)), fin);
      stv(dst, 2 * kHW + p,  clip255(g + f * (Bv - g)), fin);
    }
  }
}

__device__ void op_contrast(const Src& s, float* dst, bool fin, float f,
                            int tid, int nt, float* sred) {
  const int lane = tid & 31, wv = tid >> 5, nw = (int)(blockDim.x >> 5);
  float partial = 0.0f;
  for (int c0 = wv; c0 < kHW / 16; c0 += nw) {
    int p = c0 * 16 + (lane & 15);
    float R = 0.0f, G = 0.0f, Bv = 0.0f;
    if (lane < 16) { R = ldi(s, p); G = ldi(s, kHW + p); Bv = ldi(s, 2 * kHW + p); }
    float g = wave_gray16(R, G, Bv, lane);
    if (lane < 16) partial += g;
  }
  float mu = block_red<0>(partial, sred, tid) / (float)kHW;
  stream_range(s.p, s.sc, dst, kCHW / 4, fin, tid, nt,
               [mu, f](float v) { return clip255(mu + f * (v - mu)); });
}

__device__ void op_sharpness(const Src& s, float* dst, bool fin, float f, int tid, int nt) {
  const float w1 = 1.0f / 13.0f, w5 = 5.0f / 13.0f;
  for (int i = tid; i < kHW; i += nt) {
    int y = i / W_, x = i - y * W_;
    bool border = (y == 0) || (y == H_ - 1) || (x == 0) || (x == W_ - 1);
#pragma unroll
    for (int ch = 0; ch < C_; ++ch) {
      float vc = ldpx(s, ch, y, x);
      float o;
      if (border) {
        o = vc;  // borders restored to original
      } else {
        float acc = w1 * ldpx(s, ch, y - 1, x - 1) + w1 * ldpx(s, ch, y - 1, x)
                  + w1 * ldpx(s, ch, y - 1, x + 1) + w1 * ldpx(s, ch, y, x - 1)
                  + w5 * vc                        + w1 * ldpx(s, ch, y, x + 1)
                  + w1 * ldpx(s, ch, y + 1, x - 1) + w1 * ldpx(s, ch, y + 1, x)
                  + w1 * ldpx(s, ch, y + 1, x + 1);
        o = clip255(acc + f * (vc - acc));
      }
      stv(dst, ch * kHW + i, o, fin);
    }
  }
}

__device__ __forceinline__ int q255(float v) {
  int q = (int)rintf(v);            // RNE then clip, matching the reference
  return min(255, max(0, q));
}

__device__ void op_equalize(const Src& s, float* dst, bool fin, int tid, int nt,
                            int* hist, int* lut) {
  for (int ch = 0; ch < C_; ++ch) {
    const float4* sp = (const float4*)(s.p + ch * kHW);
    float4* dp = (float4*)(dst + ch * kHW);
    for (int i = tid; i < 256; i += nt) hist[i] = 0;
    __syncthreads();
    for (int i = tid; i < kHW / 4; i += nt) {
      float4 v = sp[i];
      atomicAdd(&hist[q255(v.x * s.sc)], 1);
      atomicAdd(&hist[q255(v.y * s.sc)], 1);
      atomicAdd(&hist[q255(v.z * s.sc)], 1);
      atomicAdd(&hist[q255(v.w * s.sc)], 1);
    }
    __syncthreads();
    if (tid == 0) {
      int step = (kHW - hist[255]) / 255;
      int cum = 0;
      for (int i = 0; i < 256; ++i) {
        int cex = cum; cum += hist[i];
        lut[i] = (step > 0) ? min(255, max(0, (cex + step / 2) / step)) : i;
      }
    }
    __syncthreads();
    for (int i = tid; i < kHW / 4; i += nt) {
      float4 v = sp[i];
      float4 o;
      o.x = (float)lut[q255(v.x * s.sc)];
      o.y = (float)lut[q255(v.y * s.sc)];
      o.z = (float)lut[q255(v.z * s.sc)];
      o.w = (float)lut[q255(v.w * s.sc)];
      dp[i] = fin4(o, fin);
    }
    __syncthreads();
  }
}

__device__ void op_autocontrast(const Src& s, float* dst, bool fin, int tid, int nt,
                                float* sred) {
  for (int ch = 0; ch < C_; ++ch) {
    const float4* sp = (const float4*)(s.p + ch * kHW);
    float lo = 3.0e38f, hi = -3.0e38f;
    for (int i = tid; i < kHW / 4; i += nt) {
      float4 v = sp[i];
      float a = v.x * s.sc, b = v.y * s.sc, c = v.z * s.sc, d = v.w * s.sc;
      lo = fminf(lo, fminf(fminf(a, b), fminf(c, d)));
      hi = fmaxf(hi, fmaxf(fmaxf(a, b), fmaxf(c, d)));
    }
    lo = block_red<1>(lo, sred, tid);
    hi = block_red<2>(hi, sred, tid);
    float scl = 255.0f / fmaxf(hi - lo, 1e-6f);
    bool stretch = hi > lo;
    float flo = lo;
    stream_range(s.p + ch * kHW, s.sc, dst + ch * kHW, kHW / 4, fin, tid, nt,
                 [stretch, flo, scl](float v) {
                   return stretch ? clip255((v - flo) * scl) : v;
                 });
  }
}

__device__ void apply_op(int op, float m, const Src& s, float* dst, bool fin,
                         int tid, int nt, float* sred, int* hist, int* lut) {
  switch (op) {
    case 0: { float v = linf(m, -0.3f, 0.3f);                      // shear_x
      op_affine(s, dst, fin, 1.0f, v, 0.0f, 0.0f, 1.0f, 0.0f, tid, nt); } break;
    case 1: { float v = linf(m, -0.3f, 0.3f);                      // shear_y
      op_affine(s, dst, fin, 1.0f, 0.0f, 0.0f, v, 1.0f, 0.0f, tid, nt); } break;
    case 2: { float v = linf(m, -0.45f, 0.45f) * (float)W_;        // translate_x
      op_affine(s, dst, fin, 1.0f, 0.0f, v, 0.0f, 1.0f, 0.0f, tid, nt); } break;
    case 3: { float v = linf(m, -0.45f, 0.45f) * (float)H_;        // translate_y
      op_affine(s, dst, fin, 1.0f, 0.0f, 0.0f, 0.0f, 1.0f, v, tid, nt); } break;
    case 4: {                                                      // rotate
      float rad = linf(m, -30.0f, 30.0f) * (float)(M_PI / 180.0);
      float cs = cosf(rad), sn = sinf(rad);
      float cx = (float)(W_ - 1) * 0.5f, cy = (float)(H_ - 1) * 0.5f;
      float cc = cx - cs * cx - sn * cy;
      float ff = cy + sn * cx - cs * cy;
      op_affine(s, dst, fin, cs, sn, cc, -sn, cs, ff, tid, nt); } break;
    case 5: { float f = linf(m, 0.1f, 1.9f);                       // brightness
      stream_range(s.p, s.sc, dst, kCHW / 4, fin, tid, nt,
                   [f](float v) { return clip255(v * f); }); } break;
    case 6: op_color(s, dst, fin, linf(m, 0.1f, 1.9f), tid); break;
    case 7: op_sharpness(s, dst, fin, linf(m, 0.1f, 1.9f), tid, nt); break;
    case 8: op_contrast(s, dst, fin, linf(m, 0.1f, 1.9f), tid, nt, sred); break;
    case 9: { float thr = linf(m, 0.0f, 256.0f);                   // solarize
      stream_range(s.p, s.sc, dst, kCHW / 4, fin, tid, nt,
                   [thr](float v) { return (v < thr) ? v : 255.0f - v; }); } break;
    case 10: { float q = exp2f(rintf(linf(m, 0.0f, 4.0f)));        // posterize
      stream_range(s.p, s.sc, dst, kCHW / 4, fin, tid, nt,
                   [q](float v) { return floorf(v / q) * q; }); } break;
    case 11: op_equalize(s, dst, fin, tid, nt, hist, lut); break;
    case 12: op_autocontrast(s, dst, fin, tid, nt, sred); break;
    default:                                                        // identity
      stream_range(s.p, s.sc, dst, kCHW / 4, fin, tid, nt,
                   [](float v) { return v; });
      break;
  }
}

// ---------------- kernels ----------------
__global__ void __launch_bounds__(256) ra_max_kernel(const float* __restrict__ x,
                                                     unsigned* gmax, int n) {
  __shared__ float sred[32];
  assume_global(x);
  float m = 0.0f;  // inputs are non-negative
  const int n4 = n >> 2;
  const float4* x4 = (const float4*)x;
  const int gid = (int)(blockIdx.x * blockDim.x + threadIdx.x);
  const int gsz = (int)(gridDim.x * blockDim.x);
  for (int i = gid; i < n4; i += gsz) {
    float4 v = x4[i];
    m = fmaxf(m, fmaxf(fmaxf(v.x, v.y), fmaxf(v.z, v.w)));
  }
  for (int i = n4 * 4 + gid; i < n; i += gsz) m = fmaxf(m, x[i]);  // tail
  m = block_red<2>(m, sred, (int)threadIdx.x);
  if (threadIdx.x == 0) atomicMax(gmax, __float_as_uint(m));  // bits-compare OK for >=0
}

__global__ void __launch_bounds__(NT_MAIN) ra_main_kernel(
    const float* __restrict__ x, const float* __restrict__ ra,
    const int* __restrict__ opi, const unsigned char* __restrict__ amask,
    const unsigned* __restrict__ gmaxbits, float* __restrict__ tmp,
    float* __restrict__ out, int nimg) {
  __shared__ float s_red[32];
  __shared__ int s_hist[256];
  __shared__ int s_lut[256];
  const int b = blockIdx.x;
  if (b >= nimg) return;
  const int tid = (int)threadIdx.x, nt = (int)blockDim.x;
  const float scale = (__uint_as_float(*gmaxbits) <= 1.0f) ? 255.0f : 1.0f;

  assume_global(x); assume_global(tmp); assume_global(out);

  // depth 0: scaled input -> tmp
  {
    int op = amask[b] ? opi[b] : 13;
    Src s{ x + (size_t)b * kCHW, scale };
    apply_op(op, ra[b], s, tmp + (size_t)b * kCHW, false, tid, nt, s_red, s_hist, s_lut);
  }
  __threadfence_block();
  __syncthreads();
  // depth 1: tmp -> out (with final /255 + clip)
  {
    int op = amask[nimg + b] ? opi[nimg + b] : 13;
    Src s{ tmp + (size_t)b * kCHW, 1.0f };
    apply_op(op, ra[nimg + b], s, out + (size_t)b * kCHW, true, tid, nt, s_red, s_hist, s_lut);
  }
}

extern "C" void kernel_launch(void* const* d_in, const int* in_sizes, int n_in,
                              void* d_out, int out_size, void* d_ws, size_t ws_size,
                              hipStream_t stream) {
  (void)n_in; (void)out_size; (void)ws_size;
  const float* x = (const float*)d_in[0];            // (B,C,H,W) f32
  const float* ra = (const float*)d_in[1];           // (DEPTH,B) f32
  const int* opi = (const int*)d_in[2];              // (DEPTH,B) i32
  // jax bool arrays are 1 byte/element; read apply_mask as raw bytes.
  const unsigned char* amask = (const unsigned char*)d_in[3];
  float* out = (float*)d_out;

  unsigned* gmax = (unsigned*)d_ws;                       // 4B scalar
  float* tmp = (float*)((char*)d_ws + 256);               // B*C*H*W f32 ping buffer

  const int n = in_sizes[0];
  const int nimg = n / kCHW;

  hipMemsetAsync(d_ws, 0, 4, stream);                     // gmax = 0 (capturable)
  ra_max_kernel<<<1024, 256, 0, stream>>>(x, gmax, n);
  ra_main_kernel<<<nimg, NT_MAIN, 0, stream>>>(x, ra, opi, amask, gmax, tmp, out, nimg);
}
